// AdjacentAttentionNetwork_11407433138538
// MI455X (gfx1250) — matrix operand order repaired and checked by reading, not compiled, not run on hardware
//
#include <hip/hip_runtime.h>
#include <cstdint>
#include <cstddef>

// ---------------------------------------------------------------------------
// AdjacentAttentionNetwork forward for MI455X (gfx1250, wave32, WMMA + TDM).
// B=2, N=4096, DIM=256, HEADS=4, DIM_HEAD=64, DEPTH=4, CUTOFF=32, FF_MULT=4.
// GEMMs: bf16 WMMA (v_wmma_f32_16x16x32_bf16) with fp32 accumulate, LDS tiles
// staged by the Tensor Data Mover (tensor_load_to_lds, TENSORcnt-tracked),
// double-buffered, 64x128 block tile, 4 WMMA per wave per K-step.
// ---------------------------------------------------------------------------

#define BB 2
#define NN 4096
#define DIM 256
#define HEADS 4
#define DH 64
#define DEPTH 4
#define CUT 32
#define ROWS (BB * NN)          // 8192
#define QKVW (3 * HEADS * DH)   // 768
#define FFW (4 * DIM)           // 1024

typedef __attribute__((ext_vector_type(16))) __bf16 v16bf;
typedef __attribute__((ext_vector_type(8)))  float  v8f;
typedef __attribute__((ext_vector_type(4)))  unsigned int u32x4;
typedef __attribute__((ext_vector_type(8)))  int         i32x8;
typedef __attribute__((ext_vector_type(4)))  int         i32x4;

union FragBF { v16bf v; unsigned int u[8]; };

__device__ __forceinline__ unsigned short f2bf(float f) {
    unsigned u = __float_as_uint(f);
    unsigned r = u + 0x7FFFu + ((u >> 16) & 1u);   // round-to-nearest-even
    return (unsigned short)(r >> 16);
}
__device__ __forceinline__ float wred_sum(float v) {
    #pragma unroll
    for (int m = 16; m > 0; m >>= 1) v += __shfl_xor(v, m, 32);
    return v;
}
__device__ __forceinline__ float wred_max(float v) {
    #pragma unroll
    for (int m = 16; m > 0; m >>= 1) v = fmaxf(v, __shfl_xor(v, m, 32));
    return v;
}
__device__ __forceinline__ float gelu_exact(float x) {
    return 0.5f * x * (1.0f + erff(x * 0.70710678118654752f));
}

// ---------------------------------------------------------------------------
// TDM 2D tile load: global (bf16, row stride `stride_elems`) -> LDS with
// 40-elem padded rows (32 data elems = 16 DWORDs, then 4 DWORDs pad).
// Descriptor packing per CDNA5 ISA 8.3/8.4 (2D: groups 2/3 zero).
// ---------------------------------------------------------------------------
__device__ __forceinline__ void tdm_load_tile(unsigned lds_off, const void* gptr,
                                              unsigned tile_d1,
                                              unsigned stride_elems) {
    unsigned long long ga = (unsigned long long)gptr;
    u32x4 g0;
    g0[0] = 1u;                                        // count=1 (user mode)
    g0[1] = lds_off;                                   // lds_addr
    g0[2] = (unsigned)(ga & 0xFFFFFFFFu);              // global_addr[31:0]
    g0[3] = (unsigned)((ga >> 32) & 0x01FFFFFFu)       // global_addr[56:32]
          | (2u << 30);                                // type=2 ("image")
    i32x8 g1;
    g1[0] = (1 << 16)        // data_size = 2 bytes (bf16)
          | (1 << 20)        // pad_enable
          | (3 << 22)        // pad_interval: 16 DWORDs of data
          | (3 << 25);       // pad_amount: 4 DWORDs of pad  -> 40-elem pitch
    g1[1] = (int)(32u << 16);                          // tensor_dim0 = 32
    g1[2] = (int)(tile_d1 << 16);                      // tensor_dim1 = tile_d1
    g1[3] = (int)(32u << 16);                          // tile_dim0 = 32
    g1[4] = (int)tile_d1;                              // tile_dim1 (tile_dim2=0)
    g1[5] = (int)stride_elems;                         // tensor_dim0_stride lo
    g1[6] = 0;
    g1[7] = 0;
    i32x4 z4 = {0, 0, 0, 0};
    i32x8 z8 = {0, 0, 0, 0, 0, 0, 0, 0};
    __builtin_amdgcn_tensor_load_to_lds(g0, g1, z4, z4, z8, 0);
}

// ---------------------------------------------------------------------------
// fp32 [K,N] -> bf16 transposed [N,K] (weight conversion), grid-stride.
// ---------------------------------------------------------------------------
__global__ void k_w_to_bf16_t(const float* __restrict__ in,
                              unsigned short* __restrict__ out, int K, int N) {
    int total = K * N;
    for (int i = blockIdx.x * blockDim.x + threadIdx.x; i < total;
         i += gridDim.x * blockDim.x) {
        int k = i / N, n = i - k * N;
        out[(size_t)n * K + k] = f2bf(in[i]);
    }
}

// ---------------------------------------------------------------------------
// top-k(32) of (adjacency | eye) per row: first 32 set columns ascending
// (JAX top_k tie-break on a 0/1 matrix). One wave per row.
// ---------------------------------------------------------------------------
__global__ void k_topk(const unsigned char* __restrict__ adj,
                       int* __restrict__ idx, int* __restrict__ cnt) {
    int wave = threadIdx.x >> 5, lane = threadIdx.x & 31;
    int row = blockIdx.x * 4 + wave;
    if (row >= ROWS) return;
    int n = row & (NN - 1);
    size_t base = (size_t)row * NN;
    unsigned lt_mask = (lane == 0) ? 0u : (0xFFFFFFFFu >> (32 - lane));
    int total = 0;
    for (int c0 = 0; c0 < NN && total < CUT; c0 += 32) {
        int col = c0 + lane;
        bool pred = (adj[base + col] != 0) || (col == n);
        unsigned mask = __builtin_amdgcn_ballot_w32(pred);
        int rank = __builtin_popcount(mask & lt_mask);
        int pos = total + rank;
        if (pred && pos < CUT) idx[(size_t)row * CUT + pos] = col;
        total += __builtin_popcount(mask);
    }
    int c = total < CUT ? total : CUT;
    if (lane >= c) idx[(size_t)row * CUT + lane] = n;   // safe padding
    if (lane == 0) cnt[row] = c;
}

// ---------------------------------------------------------------------------
// LayerNorm over DIM=256, one wave per row, bf16 output.
// ---------------------------------------------------------------------------
__global__ void k_layernorm(const float* __restrict__ x,
                            const float* __restrict__ w,
                            const float* __restrict__ b,
                            unsigned short* __restrict__ out) {
    int wave = threadIdx.x >> 5, lane = threadIdx.x & 31;
    int row = blockIdx.x * 8 + wave;
    const float* xr = x + (size_t)row * DIM;
    int c = lane * 8;
    float4 a0 = *(const float4*)(xr + c);
    float4 a1 = *(const float4*)(xr + c + 4);
    float s = a0.x + a0.y + a0.z + a0.w + a1.x + a1.y + a1.z + a1.w;
    float q = a0.x*a0.x + a0.y*a0.y + a0.z*a0.z + a0.w*a0.w
            + a1.x*a1.x + a1.y*a1.y + a1.z*a1.z + a1.w*a1.w;
    s = wred_sum(s); q = wred_sum(q);
    float mu = s * (1.0f / DIM);
    float var = q * (1.0f / DIM) - mu * mu;
    float rstd = rsqrtf(var + 1e-5f);
    unsigned short o[8];
    float vin[8] = {a0.x, a0.y, a0.z, a0.w, a1.x, a1.y, a1.z, a1.w};
    #pragma unroll
    for (int i = 0; i < 8; ++i)
        o[i] = f2bf((vin[i] - mu) * rstd * w[c + i] + b[c + i]);
    *(uint4*)(out + (size_t)row * DIM + c) = *(uint4*)o;
}

// ---------------------------------------------------------------------------
// bf16 WMMA GEMM: C[M,N] = A[M,K] * W[K,N], W supplied transposed (Wt[N,K]).
// 64x128 block tile, 8 waves; wave w -> 16-row strip (w&3), 64-col half
// (w>>2): 4 C tiles, 1 A frag + 4 B frags, 4 WMMA per 32-K step.
// LDS double-buffered; next tile staged by the TDM (wave 0 issues two
// tensor_load_to_lds; s_wait_tensorcnt 0 + barrier gate the current buffer).
// Fragment lane layout per CDNA5 ISA 7.12.2:
//   A 16x32: lane l -> M = l&15; reg j -> K = (j<4?0:16) + 8*(l>>4) + 2*(j&3)
//   B 32x16: lane l -> N = l&15; reg j -> K = (j<4?0:16) + 8*(l>>4) + 2*(j&3)
// Both collapse to two contiguous ds_load_b128 per fragment.
// EPI: 0 = fp32 store, 1 = +bias +residual fp32 store, 2 = +bias GELU bf16.
// ---------------------------------------------------------------------------
template<int EPI>
__global__ void k_gemm_wmma(const unsigned short* __restrict__ A,
                            const unsigned short* __restrict__ Wt,
                            const float* __restrict__ bias,
                            const float* __restrict__ resid,
                            void* __restrict__ outp,
                            int M, int N, int K) {
    __shared__ unsigned short As[2][64][40];    // 10 KB (80B pitch)
    __shared__ unsigned short Bs[2][128][40];   // 20 KB (Wt rows: [n][k])
    int tid = threadIdx.x;
    int wave = tid >> 5, lane = tid & 31;
    int mw = wave & 3, nhalf = wave >> 2;
    int m0 = blockIdx.y * 64, n0 = blockIdx.x * 128;
    int hi = lane >> 4, nl = lane & 15;
    v8f acc[4] = {};

    auto stage = [&](int buf, int k0) {   // wave 0 only; uniform descriptors
        tdm_load_tile((unsigned)(size_t)&As[buf][0][0],
                      A + (size_t)m0 * K + k0, 64u, (unsigned)K);
        tdm_load_tile((unsigned)(size_t)&Bs[buf][0][0],
                      Wt + (size_t)n0 * K + k0, 128u, (unsigned)K);
    };

    const int nsteps = K >> 5;
    if (wave == 0) stage(0, 0);
    int cur = 0;
    for (int s = 0; s < nsteps; ++s) {
        if (wave == 0) __builtin_amdgcn_s_wait_tensorcnt(0);
        __syncthreads();                       // buffer `cur` is ready
        if (wave == 0 && s + 1 < nsteps) stage(cur ^ 1, (s + 1) << 5);

        FragBF a, b[4];
        #pragma unroll
        for (int j = 0; j < 8; ++j) {
            int kbA = ((j < 4) ? 0 : 16) + 8 * hi + 2 * (j & 3);
            a.u[j] = *(const unsigned*)&As[cur][16 * mw + nl][kbA];
        }
        #pragma unroll
        for (int t = 0; t < 4; ++t) {
            int nb = 64 * nhalf + 16 * t + nl;
            #pragma unroll
            for (int j = 0; j < 8; ++j) {
                int kbB = ((j < 4) ? 0 : 16) + 8 * hi + 2 * (j & 3);
                b[t].u[j] = *(const unsigned*)&Bs[cur][nb][kbB];
            }
        }
        #pragma unroll
        for (int t = 0; t < 4; ++t)
            acc[t] = __builtin_amdgcn_wmma_f32_16x16x32_bf16(
                false, a.v, false, b[t].v, (short)0, acc[t], false, false);
        cur ^= 1;
    }

    // epilogue: C reg r, lanes 0-15 -> M=r, N=lane; lanes 16-31 -> M=r+8
    #pragma unroll
    for (int t = 0; t < 4; ++t) {
        #pragma unroll
        for (int r = 0; r < 8; ++r) {
            int m = m0 + 16 * mw + r + 8 * hi;
            int n = n0 + 64 * nhalf + 16 * t + nl;
            float v = acc[t][r];
            if (EPI == 0) {
                ((float*)outp)[(size_t)m * N + n] = v;
            } else if (EPI == 1) {
                ((float*)outp)[(size_t)m * N + n] =
                    v + bias[n] + resid[(size_t)m * N + n];
            } else {
                ((unsigned short*)outp)[(size_t)m * N + n] =
                    f2bf(gelu_exact(v + bias[n]));
            }
        }
    }
}

// ---------------------------------------------------------------------------
// Gather attention: one wave per (row, head). 33 keys = null + 32 gathered.
// qkv fp32 layout per row: [q(256) | k(256) | v(256)], head h at h*64.
// ---------------------------------------------------------------------------
__global__ void k_attention(const float* __restrict__ qkv,
                            const int* __restrict__ idx,
                            const int* __restrict__ cnt,
                            const float* __restrict__ null_k,
                            const float* __restrict__ null_v,
                            unsigned short* __restrict__ o_bf16) {
    __shared__ float qs[4][64];
    int wave = threadIdx.x >> 5, lane = threadIdx.x & 31;
    int gid = blockIdx.x * 4 + wave;
    int h = gid & (HEADS - 1);
    int row = gid >> 2;                 // 0..ROWS-1
    int b = row >> 12;                  // row / NN
    const float* qr = qkv + (size_t)row * QKVW + h * DH;
    int d0 = 2 * lane;
    float2 q2 = *(const float2*)(qr + d0);
    qs[wave][d0] = q2.x; qs[wave][d0 + 1] = q2.y;
    __syncthreads();

    const float scale = 0.125f;         // 64^-0.5
    const float* nk = null_k + h * DH;
    float snull = wred_sum(q2.x * nk[d0] + q2.y * nk[d0 + 1]) * scale;

    int crow = cnt[row];
    int j = idx[(size_t)row * CUT + lane];
    bool valid = lane < crow;
    const float* kp = qkv + ((size_t)b * NN + j) * QKVW + DIM + h * DH;
    float s = 0.f;
    #pragma unroll
    for (int d = 0; d < DH; d += 4) {
        float4 kk = *(const float4*)(kp + d);
        float4 qq = *(const float4*)&qs[wave][d];
        s += kk.x * qq.x + kk.y * qq.y + kk.z * qq.z + kk.w * qq.w;
    }
    s *= scale;
    float sm = valid ? s : -3.0e38f;
    float mx = fmaxf(wred_max(sm), snull);
    float e = valid ? __expf(s - mx) : 0.f;
    float enull = __expf(snull - mx);
    float denom = wred_sum(e) + enull;
    float wgt = e / denom;
    float wnull = enull / denom;

    const float* nv = null_v + h * DH;
    float ax = wnull * nv[d0], ay = wnull * nv[d0 + 1];
    #pragma unroll 4
    for (int a = 0; a < CUT; ++a) {
        float w = __shfl(wgt, a, 32);
        int jj = __shfl(j, a, 32);
        if (w > 0.f) {
            const float* vp = qkv + ((size_t)b * NN + jj) * QKVW + 2 * DIM + h * DH;
            float2 vv = *(const float2*)(vp + d0);
            ax += w * vv.x; ay += w * vv.y;
        }
    }
    unsigned short p[2] = {f2bf(ax), f2bf(ay)};
    *(unsigned*)(o_bf16 + (size_t)row * DIM + h * DH + d0) = *(unsigned*)p;
}

// ---------------------------------------------------------------------------
// host orchestration
// ---------------------------------------------------------------------------
extern "C" void kernel_launch(void* const* d_in, const int* in_sizes, int n_in,
                              void* d_out, int out_size, void* d_ws, size_t ws_size,
                              hipStream_t stream) {
    (void)in_sizes; (void)n_in; (void)out_size; (void)ws_size;
    const float* x = (const float*)d_in[0];
    const unsigned char* adj = (const unsigned char*)d_in[1];

    char* ws = (char*)d_ws;
    size_t off = 0;
    auto alloc = [&](size_t bytes) { char* p = ws + off; off += (bytes + 255) & ~(size_t)255; return p; };
    float*          xbuf  = (float*)          alloc((size_t)ROWS * DIM * 4);
    unsigned short* hbuf  = (unsigned short*) alloc((size_t)ROWS * DIM * 2);
    float*          qkv   = (float*)          alloc((size_t)ROWS * QKVW * 4);
    unsigned short* obuf  = (unsigned short*) alloc((size_t)ROWS * DIM * 2);
    unsigned short* ffh   = (unsigned short*) alloc((size_t)ROWS * FFW * 2);
    int*            idx   = (int*)            alloc((size_t)ROWS * CUT * 4);
    int*            cnt   = (int*)            alloc((size_t)ROWS * 4);
    unsigned short* wqb   = (unsigned short*) alloc((size_t)DIM * QKVW * 2);
    unsigned short* wob   = (unsigned short*) alloc((size_t)DIM * DIM * 2);
    unsigned short* wf1b  = (unsigned short*) alloc((size_t)DIM * FFW * 2);
    unsigned short* wf2b  = (unsigned short*) alloc((size_t)FFW * DIM * 2);

    (void)hipMemcpyAsync(xbuf, x, (size_t)ROWS * DIM * 4, hipMemcpyDeviceToDevice, stream);
    k_topk<<<ROWS / 4, 128, 0, stream>>>(adj, idx, cnt);

    for (int l = 0; l < DEPTH; ++l) {
        int base = 2 + 13 * l;   // setup_inputs() dict insertion order
        const float* ln1_w  = (const float*)d_in[base + 0];
        const float* ln1_b  = (const float*)d_in[base + 1];
        const float* w_qkv  = (const float*)d_in[base + 2];
        const float* w_out  = (const float*)d_in[base + 3];
        const float* b_out  = (const float*)d_in[base + 4];
        const float* null_k = (const float*)d_in[base + 5];
        const float* null_v = (const float*)d_in[base + 6];
        const float* ln2_w  = (const float*)d_in[base + 7];
        const float* ln2_b  = (const float*)d_in[base + 8];
        const float* w_ff1  = (const float*)d_in[base + 9];
        const float* b_ff1  = (const float*)d_in[base + 10];
        const float* w_ff2  = (const float*)d_in[base + 11];
        const float* b_ff2  = (const float*)d_in[base + 12];

        k_w_to_bf16_t<<<512, 256, 0, stream>>>(w_qkv, wqb, DIM, QKVW);
        k_w_to_bf16_t<<<256, 256, 0, stream>>>(w_out, wob, DIM, DIM);
        k_w_to_bf16_t<<<512, 256, 0, stream>>>(w_ff1, wf1b, DIM, FFW);
        k_w_to_bf16_t<<<512, 256, 0, stream>>>(w_ff2, wf2b, FFW, DIM);

        // attention branch
        k_layernorm<<<ROWS / 8, 256, 0, stream>>>(xbuf, ln1_w, ln1_b, hbuf);
        k_gemm_wmma<0><<<dim3(QKVW / 128, ROWS / 64), 256, 0, stream>>>(
            hbuf, wqb, nullptr, nullptr, qkv, ROWS, QKVW, DIM);
        k_attention<<<ROWS * HEADS / 4, 128, 0, stream>>>(qkv, idx, cnt, null_k, null_v, obuf);
        k_gemm_wmma<1><<<dim3(DIM / 128, ROWS / 64), 256, 0, stream>>>(
            obuf, wob, b_out, xbuf, xbuf, ROWS, DIM, DIM);

        // feedforward branch
        k_layernorm<<<ROWS / 8, 256, 0, stream>>>(xbuf, ln2_w, ln2_b, hbuf);
        k_gemm_wmma<2><<<dim3(FFW / 128, ROWS / 64), 256, 0, stream>>>(
            hbuf, wf1b, b_ff1, nullptr, ffh, ROWS, FFW, DIM);
        k_gemm_wmma<1><<<dim3(DIM / 128, ROWS / 64), 256, 0, stream>>>(
            ffh, wf2b, b_ff2, xbuf, xbuf, ROWS, DIM, FFW);
    }

    (void)hipMemcpyAsync(d_out, xbuf, (size_t)ROWS * DIM * 4, hipMemcpyDeviceToDevice, stream);
}
